// LayerNormInputBinaryTreeLSTMCell_61366492725519
// MI455X (gfx1250) — matrix operand order repaired and checked by reading, not compiled, not run on hardware
//
#include <hip/hip_runtime.h>

#define H 128
#define EPS 1e-5f
#define ASTRIDE 264   // 16-byte-aligned, shifts banks by 4 per row -> conflict-free b128 LDS reads

typedef _Float16 h4_t  __attribute__((ext_vector_type(4)));
typedef _Float16 h8_t  __attribute__((ext_vector_type(8)));
typedef _Float16 h16_t __attribute__((ext_vector_type(16)));
typedef float    v8f   __attribute__((ext_vector_type(8)));

// ---------------------------------------------------------------------------
// Hardware transcendental tanh (CDNA5 V_TANH_F32), with fallbacks.
// ---------------------------------------------------------------------------
__device__ __forceinline__ float hw_tanh(float x) {
#if __has_builtin(__builtin_amdgcn_tanhf)
  return __builtin_amdgcn_tanhf(x);
#elif __has_builtin(__builtin_amdgcn_tanh_f32)
  return __builtin_amdgcn_tanh_f32(x);
#else
  float r;
  asm("v_tanh_f32 %0, %1" : "=v"(r) : "v"(x));
  return r;
#endif
}
// sigmoid(x) = 0.5 + 0.5*tanh(0.5*x) : mul + tanh + fma
__device__ __forceinline__ float hw_sigmoid(float x) {
  return fmaf(0.5f, hw_tanh(0.5f * x), 0.5f);
}

// Butterfly sum over the 16-lane half-wave (xor 1,2,4,8 never crosses halves).
// Single ds_swizzle_b32 per step: offset = (xor_mask<<10) | and_mask(0x1f).
__device__ __forceinline__ float hred16(float x) {
  x += __int_as_float(__builtin_amdgcn_ds_swizzle(__float_as_int(x), 0x041F));
  x += __int_as_float(__builtin_amdgcn_ds_swizzle(__float_as_int(x), 0x081F));
  x += __int_as_float(__builtin_amdgcn_ds_swizzle(__float_as_int(x), 0x101F));
  x += __int_as_float(__builtin_amdgcn_ds_swizzle(__float_as_int(x), 0x201F));
  return x;
}

// ---------------------------------------------------------------------------
// Pack [Wl;Wr] (256 x 640 f32, row-major) into f16 B-fragment order:
// Wp[((t*8 + s)*32 + lane)*16 + h] = W[32*s + 8*(lane>>4) + (h&7) + 16*(h>>3)][16*t + (lane&15)]
// ---------------------------------------------------------------------------
__global__ void pack_weights(const float* __restrict__ Wl,
                             const float* __restrict__ Wr,
                             _Float16* __restrict__ Wp) {
  int idx = blockIdx.x * 256 + threadIdx.x;
  if (idx >= 256 * 640) return;
  int h    = idx & 15;
  int lane = (idx >> 4) & 31;
  int s    = (idx >> 9) & 7;
  int t    = idx >> 12;                    // 0..39
  int k = 32 * s + 8 * (lane >> 4) + (h & 7) + 16 * (h >> 3);   // 0..255
  int n = 16 * t + (lane & 15);            // 0..639
  float w = (k < 128) ? Wl[k * 640 + n] : Wr[(k - 128) * 640 + n];
  Wp[idx] = (_Float16)w;
}

// ---------------------------------------------------------------------------
// Fused TreeLSTM cell: GEMM (f16 WMMA, f32 accum) + 6x LayerNorm + epilogue.
// 128 threads = 4 waves; wave handles 16 rows; block handles 64 rows.
// ---------------------------------------------------------------------------
__launch_bounds__(128, 1)
__global__ void treelstm_ln_kernel(
    const float* __restrict__ lhidden, const float* __restrict__ rhidden,
    const float* __restrict__ lcell,   const float* __restrict__ rcell,
    const _Float16* __restrict__ Wp,   const float* __restrict__ br,
    const float* __restrict__ g_u,  const float* __restrict__ b_u,
    const float* __restrict__ g_i,  const float* __restrict__ b_i,
    const float* __restrict__ g_o,  const float* __restrict__ b_o,
    const float* __restrict__ g_lf, const float* __restrict__ b_lf,
    const float* __restrict__ g_rf, const float* __restrict__ b_rf,
    const float* __restrict__ g_c,  const float* __restrict__ b_c,
    float* __restrict__ out, long Ntot) {

  // per-wave 8448B region: f16 A-staging first, then reused as f32 act buffer
  __shared__ __align__(32) unsigned char smem[4 * (ASTRIDE * 16 * 2)];
  __shared__ float gp[6 * 256];   // [gate][gamma 0..127 | beta 128..255]
  __shared__ float brS[640];

  const int tid  = threadIdx.x;
  const int lane = tid & 31;
  const int wave = tid >> 5;
  const int nn   = lane & 15;     // N-column within 16-wide tile / C-frag column
  const int hl   = lane >> 4;     // half-wave select

  {                               // blockDim.x == 128, no guard needed
    gp[tid]        = g_u[tid];  gp[128  + tid] = b_u[tid];
    gp[256  + tid] = g_i[tid];  gp[384  + tid] = b_i[tid];
    gp[512  + tid] = g_o[tid];  gp[640  + tid] = b_o[tid];
    gp[768  + tid] = g_lf[tid]; gp[896  + tid] = b_lf[tid];
    gp[1024 + tid] = g_rf[tid]; gp[1152 + tid] = b_rf[tid];
    gp[1280 + tid] = g_c[tid];  gp[1408 + tid] = b_c[tid];
  }
  for (int i = tid; i < 640; i += 128) brS[i] = br[i];

  const long row0 = (long)blockIdx.x * 64 + wave * 16;
  _Float16* Ast  = (_Float16*)(smem + wave * (ASTRIDE * 16 * 2));
  float*    actB = (float*)   (smem + wave * (ASTRIDE * 16 * 2));

  // ---- stage 16 rows of concat(lhidden,rhidden) as f16 (row-major, padded) --
  for (int j = lane; j < 16 * 64; j += 32) {
    int r = j >> 6;                 // row 0..15
    int q = j & 63;                 // float4 index within 256-wide concat row
    const float* src = (q < 32) ? (lhidden + (row0 + r) * H + q * 4)
                                : (rhidden + (row0 + r) * H + (q - 32) * 4);
    float4 v = *(const float4*)src;
    h4_t hv = { (_Float16)v.x, (_Float16)v.y, (_Float16)v.z, (_Float16)v.w };
    *(h4_t*)(Ast + r * ASTRIDE + q * 4) = hv;
  }

  // ---- build all 8 A fragments once (reused across 40 column tiles) --------
  // lane l, row M = l&15; lanes 0-15: K = 32s+{0..7, 16..23}; lanes 16-31: +8
  h16_t a[8];
  {
    const _Float16* base = Ast + nn * ASTRIDE + 8 * hl;
#pragma unroll
    for (int s = 0; s < 8; ++s) {
      h8_t c0 = *(const h8_t*)(base + 32 * s);
      h8_t c1 = *(const h8_t*)(base + 32 * s + 16);
      a[s] = __builtin_shufflevector(c0, c1,
               0,1,2,3,4,5,6,7,8,9,10,11,12,13,14,15);
    }
  }
  __syncthreads();  // params ready; fences f16 reads before f32 actB reuse

  v8f cell[8];      // running cell accumulator, frag layout (f = col tile)

#pragma unroll
  for (int g = 0; g < 5; ++g) {          // gates: u, i, o, lf, rf
    // ---- GEMM: s outer / f inner => 8 independent accumulation chains,
    //      dependent WMMAs are 8 instructions apart (pipelines at 1 wave/SIMD)
    v8f acc[8];
#pragma unroll
    for (int f = 0; f < 8; ++f) acc[f] = (v8f){};
    const _Float16* wg = Wp + (size_t)(g * 64) * 512 + lane * 16;
#pragma unroll
    for (int s = 0; s < 8; ++s) {
#pragma unroll
      for (int f = 0; f < 8; ++f) {
        h16_t b = *(const h16_t*)(wg + (size_t)(f * 8 + s) * 512);
        acc[f] = __builtin_amdgcn_wmma_f32_16x16x32_f16(false, a[s], false, b,
                                                        (short)0, acc[f],
                                                        false, false);
      }
    }
    // ---- bias + LayerNorm statistics (row m = v + 8*hl, in-register) ------
    float bias[8], ga[8], be[8];
#pragma unroll
    for (int f = 0; f < 8; ++f) {
      bias[f] = brS[g * 128 + f * 16 + nn];
      ga[f]   = gp[g * 256 + f * 16 + nn];
      be[f]   = gp[g * 256 + 128 + f * 16 + nn];
    }
    float s1[8], s2[8];
#pragma unroll
    for (int v = 0; v < 8; ++v) { s1[v] = 0.f; s2[v] = 0.f; }
#pragma unroll
    for (int f = 0; f < 8; ++f) {
#pragma unroll
      for (int v = 0; v < 8; ++v) {
        float x = acc[f][v] + bias[f];
        acc[f][v] = x;
        s1[v] += x;
        s2[v] = fmaf(x, x, s2[v]);
      }
    }
    float rs[8], nb[8];
#pragma unroll
    for (int v = 0; v < 8; ++v) {
      float t1 = hred16(s1[v]);
      float t2 = hred16(s2[v]);
      float m   = t1 * (1.f / 128.f);
      float var = fmaf(-m, m, t2 * (1.f / 128.f));
      float r   = __frsqrt_rn(var + EPS);
      rs[v] = r;
      nb[v] = -m * r;
    }
    // ---- normalize (2 FMA), activate (hw tanh), combine --------------------
#pragma unroll
    for (int f = 0; f < 8; ++f) {
#pragma unroll
      for (int v = 0; v < 8; ++v) {
        int col = f * 16 + nn;
        int m   = v + 8 * hl;
        float z = fmaf(acc[f][v], rs[v], nb[v]);
        float y = fmaf(z, ga[f], be[f]);
        if (g == 0) {                       // u = tanh -> stash in LDS
          actB[m * H + col] = hw_tanh(y);
        } else if (g == 1) {                // i: cell = i*u
          cell[f][v] = hw_sigmoid(y) * actB[m * H + col];
        } else if (g == 2) {                // o -> stash (u is dead)
          actB[m * H + col] = hw_sigmoid(y);
        } else if (g == 3) {                // lf: cell += lf*lcell
          cell[f][v] = fmaf(hw_sigmoid(y), lcell[(row0 + m) * H + col],
                            cell[f][v]);
        } else {                            // rf: cell += rf*rcell
          cell[f][v] = fmaf(hw_sigmoid(y), rcell[(row0 + m) * H + col],
                            cell[f][v]);
        }
      }
    }
  }

  // ---- final cell LayerNorm + epilogue -------------------------------------
  {
    float ga[8], be[8];
#pragma unroll
    for (int f = 0; f < 8; ++f) {
      ga[f] = gp[5 * 256 + f * 16 + nn];
      be[f] = gp[5 * 256 + 128 + f * 16 + nn];
    }
    float s1[8], s2[8];
#pragma unroll
    for (int v = 0; v < 8; ++v) { s1[v] = 0.f; s2[v] = 0.f; }
#pragma unroll
    for (int f = 0; f < 8; ++f)
#pragma unroll
      for (int v = 0; v < 8; ++v) {
        float x = cell[f][v];
        s1[v] += x;
        s2[v] = fmaf(x, x, s2[v]);
      }
    float rs[8], nb[8];
#pragma unroll
    for (int v = 0; v < 8; ++v) {
      float t1 = hred16(s1[v]);
      float t2 = hred16(s2[v]);
      float m   = t1 * (1.f / 128.f);
      float var = fmaf(-m, m, t2 * (1.f / 128.f));
      float r   = __frsqrt_rn(var + EPS);
      rs[v] = r;
      nb[v] = -m * r;
    }
    float* outH = out;
    float* outC = out + (size_t)Ntot * H;
#pragma unroll
    for (int f = 0; f < 8; ++f) {
#pragma unroll
      for (int v = 0; v < 8; ++v) {
        int col = f * 16 + nn;
        int m   = v + 8 * hl;
        float z  = fmaf(cell[f][v], rs[v], nb[v]);
        float nc = fmaf(z, ga[f], be[f]);
        float nh = actB[m * H + col] * hw_tanh(nc);   // o * tanh(new_cell)
        outH[(row0 + m) * H + col] = nh;
        outC[(row0 + m) * H + col] = nc;
      }
    }
  }
}

// ---------------------------------------------------------------------------
extern "C" void kernel_launch(void* const* d_in, const int* in_sizes, int n_in,
                              void* d_out, int out_size, void* d_ws, size_t ws_size,
                              hipStream_t stream) {
  const float* lhidden = (const float*)d_in[0];
  const float* lcell   = (const float*)d_in[1];
  const float* rhidden = (const float*)d_in[2];
  const float* rcell   = (const float*)d_in[3];
  const float* Wl      = (const float*)d_in[4];
  const float* Wr      = (const float*)d_in[5];
  const float* br      = (const float*)d_in[6];
  const float* g_u  = (const float*)d_in[7];
  const float* b_u  = (const float*)d_in[8];
  const float* g_i  = (const float*)d_in[9];
  const float* b_i  = (const float*)d_in[10];
  const float* g_o  = (const float*)d_in[11];
  const float* b_o  = (const float*)d_in[12];
  const float* g_lf = (const float*)d_in[13];
  const float* b_lf = (const float*)d_in[14];
  const float* g_rf = (const float*)d_in[15];
  const float* b_rf = (const float*)d_in[16];
  const float* g_c  = (const float*)d_in[17];
  const float* b_c  = (const float*)d_in[18];

  _Float16* Wp = (_Float16*)d_ws;                 // 256*640*2 = 320 KB
  long Ntot = in_sizes[0] / H;                    // 131072

  pack_weights<<<(256 * 640 + 255) / 256, 256, 0, stream>>>(Wl, Wr, Wp);

  treelstm_ln_kernel<<<(int)(Ntot / 64), 128, 0, stream>>>(
      lhidden, rhidden, lcell, rcell, Wp, br,
      g_u, b_u, g_i, b_i, g_o, b_o, g_lf, b_lf, g_rf, b_rf, g_c, b_c,
      (float*)d_out, Ntot);
}
